// MultiHeadAttention_14302241096252
// MI455X (gfx1250) — matrix operand order repaired and checked by reading, not compiled, not run on hardware
//
#include <hip/hip_runtime.h>
#include <hip/hip_bf16.h>
#include <math.h>

// ---------------- problem constants ----------------
static constexpr int BB = 4;
static constexpr int SS = 2048;
static constexpr int DD = 1024;
static constexpr int HH = 16;
static constexpr int HD = 64;

typedef __bf16 bf16_t;
typedef bf16_t v16bf __attribute__((ext_vector_type(16)));
typedef bf16_t v8bf  __attribute__((ext_vector_type(8)));
typedef bf16_t v4bf  __attribute__((ext_vector_type(4)));
typedef float  v8f   __attribute__((ext_vector_type(8)));

__device__ __forceinline__ v16bf cat8(v8bf lo, v8bf hi) {
    return __builtin_shufflevector(lo, hi, 0, 1, 2, 3, 4, 5, 6, 7,
                                            8, 9, 10, 11, 12, 13, 14, 15);
}

__device__ __forceinline__ v8f wmma_bf16(v16bf a, v16bf b, v8f c) {
    // D = A(16x32 bf16) * B(32x16 bf16) + C(16x16 f32)
    return __builtin_amdgcn_wmma_f32_16x16x32_bf16(
        /*neg_a=*/false, a, /*neg_b=*/false, b,
        /*c_mod=*/(short)0, c, /*reuse_a=*/false, /*reuse_b=*/false);
}

// VALU lane-permute (v_permlane16_b32) based XOR shuffle within 16-lane halves:
// avoids ds_bpermute round trips + s_wait_dscnt in the softmax reduction.
__device__ __forceinline__ float permlane16_f(float v, int s0, int s1) {
    int x = __builtin_bit_cast(int, v);
    int p = __builtin_amdgcn_permlane16(x, x, s0, s1, false, false);
    return __builtin_bit_cast(float, p);
}
__device__ __forceinline__ float rowmax16(float v) {
    v = fmaxf(v, permlane16_f(v, 0x67452301, 0xEFCDAB89));  // xor 1
    v = fmaxf(v, permlane16_f(v, 0x54761032, 0xDCFE98BA));  // xor 2
    v = fmaxf(v, permlane16_f(v, 0x32107654, 0xBA98FEDC));  // xor 4
    v = fmaxf(v, permlane16_f(v, 0xFEDCBA98, 0x76543210));  // xor 8
    return v;
}

// ---------------- 1) conversions ----------------

__global__ void x_cvt_kernel(const float* __restrict__ X, bf16_t* __restrict__ Xb, int n4) {
    int id = blockIdx.x * blockDim.x + threadIdx.x;
    if (id >= n4) return;
    float4 f = ((const float4*)X)[id];
    v4bf o;
    o[0] = (bf16_t)f.x; o[1] = (bf16_t)f.y; o[2] = (bf16_t)f.z; o[3] = (bf16_t)f.w;
    ((v4bf*)Xb)[id] = o;
}

// W (f32, DxD row-major) -> Wt (bf16, transposed: Wt[n*D+k] = W[k*D+n])
__global__ void wt_cvt_kernel(const float* __restrict__ W, bf16_t* __restrict__ Wt) {
    int id = blockIdx.x * blockDim.x + threadIdx.x;   // 0 .. D*D-1
    int n = id & (DD - 1);
    int k = id >> 10;
    Wt[(size_t)n * DD + k] = (bf16_t)W[(size_t)k * DD + n];
}

// ---------------- 2) QKV projection GEMM ----------------
// One wave computes a 16(M) x 64(N) tile of Y = Xb @ W.
// VT=false -> write [B,H,S,HD]  (Q, K)
// VT=true  -> write [B,H,HD,S]  (V transposed for the P@V WMMA B-operand)
template <bool VT>
__global__ __launch_bounds__(256) void qkv_gemm_kernel(const bf16_t* __restrict__ X,
                                                       const bf16_t* __restrict__ Wt,
                                                       bf16_t* __restrict__ Out) {
    const int lane = threadIdx.x & 31;
    const int wid  = blockIdx.x * (blockDim.x >> 5) + (threadIdx.x >> 5);
    const int rowTile = wid >> 4;          // (B*S/16) = 512 row tiles
    const int colGrp  = wid & 15;          // 16 groups of 64 columns
    const int row0 = rowTile * 16;
    const int m = lane & 15;               // A row / B column within tile
    const int h = lane >> 4;               // lane half

    v8f acc[4] = {};
    const bf16_t* xrow = X + (size_t)(row0 + m) * DD;

    for (int kk = 0; kk < DD / 32; ++kk) {
        const int k0 = kk * 32;
        v16bf a = cat8(*(const v8bf*)(xrow + k0 + 8 * h),
                       *(const v8bf*)(xrow + k0 + 16 + 8 * h));
#pragma unroll
        for (int t = 0; t < 4; ++t) {
            const int col = colGrp * 64 + t * 16 + m;
            v16bf b = cat8(*(const v8bf*)(Wt + (size_t)col * DD + k0 + 8 * h),
                           *(const v8bf*)(Wt + (size_t)col * DD + k0 + 16 + 8 * h));
            acc[t] = wmma_bf16(a, b, acc[t]);
        }
    }

    // C layout: element r -> (M = r + 8*h, N = lane&15)
#pragma unroll
    for (int t = 0; t < 4; ++t) {
#pragma unroll
        for (int r = 0; r < 8; ++r) {
            const int orow = row0 + r + 8 * h;         // global row in [0, B*S)
            const int col  = colGrp * 64 + t * 16 + m; // model dim
            const int bidx = orow >> 11;               // / S
            const int s    = orow & (SS - 1);
            const int hh   = col >> 6;                 // head
            const int hd   = col & (HD - 1);
            size_t idx = VT
                ? ((((size_t)bidx * HH + hh) * HD + hd) * SS + s)
                : ((((size_t)bidx * HH + hh) * SS + s) * HD + hd);
            Out[idx] = (bf16_t)acc[t][r];
        }
    }
}

// ---------------- 3) flash attention ----------------
// One 32-key block step. MASKED instantiation applies the causal mask; the
// unmasked instantiation is used for all fully-in-range blocks.
template <bool MASKED>
__device__ __forceinline__ void attn_block(int kb, int qbase, int h2, int n,
                                           const bf16_t* __restrict__ Kh,
                                           const bf16_t* __restrict__ Vh,
                                           bf16_t* __restrict__ pb,
                                           const v16bf (&qa)[2], v16bf onesb,
                                           float (&mrow)[8], v8f (&oacc)[4], v8f& lacc) {
    const float sm_scale = 0.125f;           // 1/sqrt(HD)
    const float log2e    = 1.44269504088896f;

    // scores: two 16x16 subtiles over 32 keys; HD=64 contraction -> 2 wmma each
    v8f s01[2];
#pragma unroll
    for (int sub = 0; sub < 2; ++sub) {
        v8f s = {};
        const int ks = kb + sub * 16;
#pragma unroll
        for (int c = 0; c < 2; ++c) {
            v16bf bk = cat8(*(const v8bf*)(Kh + (size_t)(ks + n) * HD + c * 32 + 8 * h2),
                            *(const v8bf*)(Kh + (size_t)(ks + n) * HD + c * 32 + 16 + 8 * h2));
            s = wmma_bf16(qa[c], bk, s);
        }
        s01[sub] = s;
    }

    // (mask then) scale, matching the reference order
#pragma unroll
    for (int sub = 0; sub < 2; ++sub) {
#pragma unroll
        for (int r = 0; r < 8; ++r) {
            float v = s01[sub][r];
            if (MASKED) {
                const int qi = qbase + r + 8 * h2;
                const int ki = kb + sub * 16 + n;
                v = (ki > qi) ? -3.0e38f : v;
            }
            s01[sub][r] = v * sm_scale;
        }
    }

    // online softmax: row max via v_permlane16 within 16-lane halves; row sums
    // come from an extra WMMA against an all-ones B operand (lacc).
    float mnew[8], alpha[8];
#pragma unroll
    for (int r = 0; r < 8; ++r) {
        float v = rowmax16(fmaxf(s01[0][r], s01[1][r]));
        mnew[r]  = fmaxf(mrow[r], v);
        alpha[r] = __builtin_exp2f((mrow[r] - mnew[r]) * log2e);
        mrow[r]  = mnew[r];
    }
#pragma unroll
    for (int r = 0; r < 8; ++r) {
        s01[0][r] = __builtin_exp2f((s01[0][r] - mnew[r]) * log2e);
        s01[1][r] = __builtin_exp2f((s01[1][r] - mnew[r]) * log2e);
        lacc[r] *= alpha[r];
#pragma unroll
        for (int t = 0; t < 4; ++t) oacc[t][r] *= alpha[r];
    }

    // re-layout P (C layout -> A layout) through per-wave LDS; intra-wave only,
    // DS ops are in-order per wave so no workgroup barrier (trip counts diverge).
#pragma unroll
    for (int sub = 0; sub < 2; ++sub)
#pragma unroll
        for (int r = 0; r < 8; ++r)
            pb[(r + 8 * h2) * 32 + sub * 16 + n] = (bf16_t)s01[sub][r];
    asm volatile("s_wait_dscnt 0" ::: "memory");
    v16bf pa = cat8(*(const v8bf*)(pb + n * 32 + 8 * h2),
                    *(const v8bf*)(pb + n * 32 + 16 + 8 * h2));

    // l += rowsum(P) (replicated across N by ones-B); ctx += P @ V
    lacc = wmma_bf16(pa, onesb, lacc);
#pragma unroll
    for (int t = 0; t < 4; ++t) {
        v16bf bv = cat8(*(const v8bf*)(Vh + (size_t)(t * 16 + n) * SS + kb + 8 * h2),
                        *(const v8bf*)(Vh + (size_t)(t * 16 + n) * SS + kb + 16 + 8 * h2));
        oacc[t] = wmma_bf16(pa, bv, oacc[t]);
    }
}

// One wave handles one (b, h, 16-query tile); loops keys 32 at a time.
// Q,K: [B,H,S,HD] bf16.  Vt: [B,H,HD,S] bf16.  Ctx out: [B,S,D] bf16.
__global__ __launch_bounds__(256) void attn_kernel(const bf16_t* __restrict__ Q,
                                                   const bf16_t* __restrict__ K,
                                                   const bf16_t* __restrict__ Vt,
                                                   bf16_t* __restrict__ Ctx) {
    __shared__ __align__(64) bf16_t pbuf[8][16 * 32];  // per-wave P staging (1 KB)

    const int lane  = threadIdx.x & 31;
    const int wslot = threadIdx.x >> 5;
    const int wid   = blockIdx.x * 8 + wslot;
    const int qt    = wid & 127;           // S/16 = 128 query tiles
    const int bh    = wid >> 7;            // b*H + h  (64 total)
    const int qbase = qt * 16;
    const int h2 = lane >> 4;
    const int n  = lane & 15;

    const bf16_t* Qh = Q  + (size_t)bh * SS * HD;
    const bf16_t* Kh = K  + (size_t)bh * SS * HD;
    const bf16_t* Vh = Vt + (size_t)bh * HD * SS;
    bf16_t* pb = pbuf[wslot];

    // Q A-operand (16x64 bf16 = two K=32 chunks), kept in registers
    v16bf qa[2];
#pragma unroll
    for (int c = 0; c < 2; ++c) {
        const bf16_t* qp = Qh + (size_t)(qbase + n) * HD + c * 32 + 8 * h2;
        qa[c] = cat8(*(const v8bf*)qp, *(const v8bf*)(qp + 16));
    }

    v16bf onesb;
#pragma unroll
    for (int i = 0; i < 16; ++i) onesb[i] = (bf16_t)1.0f;

    float mrow[8];
    v8f oacc[4] = {};
    v8f lacc = {};
#pragma unroll
    for (int r = 0; r < 8; ++r) mrow[r] = -3.0e38f;

    // blocks with kb+31 <= qbase need no causal mask; exactly one masked tail.
    const int nfull = (qbase + 1) >> 5;
    for (int kb = 0; kb < nfull * 32; kb += 32)
        attn_block<false>(kb, qbase, h2, n, Kh, Vh, pb, qa, onesb, mrow, oacc, lacc);
    attn_block<true>(nfull * 32, qbase, h2, n, Kh, Vh, pb, qa, onesb, mrow, oacc, lacc);

    // epilogue: normalize and store ctx as [B,S,D] bf16
    const int b  = bh >> 4;
    const int hh = bh & 15;
    float rinv[8];
#pragma unroll
    for (int r = 0; r < 8; ++r) rinv[r] = 1.0f / lacc[r];
#pragma unroll
    for (int t = 0; t < 4; ++t) {
#pragma unroll
        for (int r = 0; r < 8; ++r) {
            const float val = oacc[t][r] * rinv[r];
            const int orow = qbase + r + 8 * h2;
            const int col  = hh * HD + t * 16 + n;
            Ctx[((size_t)b * SS + orow) * DD + col] = (bf16_t)val;
        }
    }
}

// ---------------- 4) output projection ----------------
// out = ctx @ W_o + b_o   (fp32 out). Same 16x64-per-wave tiling.
__global__ __launch_bounds__(256) void out_gemm_kernel(const bf16_t* __restrict__ Ctx,
                                                       const bf16_t* __restrict__ Wot,
                                                       const float* __restrict__ bias,
                                                       float* __restrict__ Out) {
    const int lane = threadIdx.x & 31;
    const int wid  = blockIdx.x * (blockDim.x >> 5) + (threadIdx.x >> 5);
    const int rowTile = wid >> 4;
    const int colGrp  = wid & 15;
    const int row0 = rowTile * 16;
    const int m = lane & 15;
    const int h = lane >> 4;

    v8f acc[4] = {};
    const bf16_t* crow = Ctx + (size_t)(row0 + m) * DD;

    for (int kk = 0; kk < DD / 32; ++kk) {
        const int k0 = kk * 32;
        v16bf a = cat8(*(const v8bf*)(crow + k0 + 8 * h),
                       *(const v8bf*)(crow + k0 + 16 + 8 * h));
#pragma unroll
        for (int t = 0; t < 4; ++t) {
            const int col = colGrp * 64 + t * 16 + m;
            v16bf b = cat8(*(const v8bf*)(Wot + (size_t)col * DD + k0 + 8 * h),
                           *(const v8bf*)(Wot + (size_t)col * DD + k0 + 16 + 8 * h));
            acc[t] = wmma_bf16(a, b, acc[t]);
        }
    }

#pragma unroll
    for (int t = 0; t < 4; ++t) {
#pragma unroll
        for (int r = 0; r < 8; ++r) {
            const int orow = row0 + r + 8 * h;
            const int col  = colGrp * 64 + t * 16 + m;
            Out[(size_t)orow * DD + col] = acc[t][r] + bias[col];
        }
    }
}

// ---------------- host launcher ----------------
extern "C" void kernel_launch(void* const* d_in, const int* in_sizes, int n_in,
                              void* d_out, int out_size, void* d_ws, size_t ws_size,
                              hipStream_t stream) {
    const float* x   = (const float*)d_in[0];
    const float* W_q = (const float*)d_in[1];
    const float* W_k = (const float*)d_in[2];
    const float* W_v = (const float*)d_in[3];
    const float* W_o = (const float*)d_in[4];
    const float* b_o = (const float*)d_in[5];
    float* out = (float*)d_out;

    const size_t BSD = (size_t)BB * SS * DD;   // 8,388,608
    const size_t WSZ = (size_t)DD * DD;        // 1,048,576

    // workspace layout (bf16 elements): ~88 MiB total
    bf16_t* xb  = (bf16_t*)d_ws;
    bf16_t* wqt = xb  + BSD;
    bf16_t* wkt = wqt + WSZ;
    bf16_t* wvt = wkt + WSZ;
    bf16_t* wot = wvt + WSZ;
    bf16_t* Qb  = wot + WSZ;     // [B,H,S,HD]
    bf16_t* Kb  = Qb  + BSD;     // [B,H,S,HD]
    bf16_t* Vb  = Kb  + BSD;     // [B,H,HD,S]
    bf16_t* Cb  = Vb  + BSD;     // [B,S,D]

    const dim3 blk(256);

    // 1) conversions
    x_cvt_kernel<<<(unsigned)((BSD / 4 + 255) / 256), blk, 0, stream>>>(x, xb, (int)(BSD / 4));
    wt_cvt_kernel<<<(unsigned)((WSZ + 255) / 256), blk, 0, stream>>>(W_q, wqt);
    wt_cvt_kernel<<<(unsigned)((WSZ + 255) / 256), blk, 0, stream>>>(W_k, wkt);
    wt_cvt_kernel<<<(unsigned)((WSZ + 255) / 256), blk, 0, stream>>>(W_v, wvt);
    wt_cvt_kernel<<<(unsigned)((WSZ + 255) / 256), blk, 0, stream>>>(W_o, wot);

    // 2) QKV projections: (B*S/16)*(D/64) = 8192 waves -> 1024 blocks of 8 waves
    qkv_gemm_kernel<false><<<1024, blk, 0, stream>>>(xb, wqt, Qb);
    qkv_gemm_kernel<false><<<1024, blk, 0, stream>>>(xb, wkt, Kb);
    qkv_gemm_kernel<true ><<<1024, blk, 0, stream>>>(xb, wvt, Vb);

    // 3) flash attention: B*H*(S/16) = 8192 waves -> 1024 blocks
    attn_kernel<<<1024, blk, 0, stream>>>(Qb, Kb, Vb, Cb);

    // 4) output projection
    out_gemm_kernel<<<1024, blk, 0, stream>>>(Cb, wot, b_o, out);
}
